// MSDeformableAttention_75179107549552
// MI455X (gfx1250) — matrix operand order repaired and checked by reading, not compiled, not run on hardware
//
#include <hip/hip_runtime.h>
#include <hip/hip_bf16.h>

typedef __attribute__((ext_vector_type(16))) _Float16     v16h;
typedef __attribute__((ext_vector_type(8)))  float        v8f;
typedef __attribute__((ext_vector_type(4)))  unsigned int u32x4;
typedef __attribute__((ext_vector_type(4)))  int          i32x4;
typedef __attribute__((ext_vector_type(8)))  int          i32x8;

#define BS_     32
#define LQ_     300
#define LV_     13294   // 100*100 + 50*50 + 25*25 + 13*13

// ---------------- WMMA GEMM: C[M,N] = f16(A[M,K]) x Bt16[N,K]^T + bias[N] ----------------
// Block: 256 threads = 8 waves. Block tile 64 x N(<=256), wave tile 32x64 (2x4 WMMA 16x16), BK=32.
// B tile (N x 32 halfs, padded LDS stride 40) is DMA'd by the Tensor Data Mover each k-step.
#define BM    64
#define BNMAX 256
#define BK    32
#define ASTR  40   // LDS row stride (halfs), padded vs 32
#define BSTR  40   // LDS row stride (halfs); TDM pad: 16 DWORDs data + 4 DWORDs pad = 40 halfs

template<typename OutT>
__global__ __launch_bounds__(256) void gemm_f16wmma(
    const float* __restrict__ A, const _Float16* __restrict__ Bt,  // Bt = f16 W^T, [N][K] row-major
    const float* __restrict__ bias, OutT* __restrict__ C,
    int M, int N, int K)
{
    __shared__ __attribute__((aligned(16))) _Float16 As[BM * ASTR];
    __shared__ __attribute__((aligned(16))) _Float16 Bs[BNMAX * BSTR];

    const int tid  = threadIdx.x;
    const int lane = tid & 31;
    const int wave = tid >> 5;
    const int hl   = lane & 15;
    const int hi   = lane >> 4;
    const int wm   = (wave >> 2) * 32;   // 2 wave rows in M
    const int wn   = (wave & 3) * 64;    // 4 wave cols in N
    const int m_block = blockIdx.x * BM;

    v8f c[2][4] = {};

    for (int k0 = 0; k0 < K; k0 += BK) {
        // ---- issue TDM load of B tile: N rows x 32 halfs from Bt[0..N-1][k0..k0+31] -> Bs[n][k] (stride 40)
        if (wave == 0) {
            unsigned lds_base = (unsigned)(uintptr_t)(&Bs[0]);
            unsigned long long ga = (unsigned long long)(uintptr_t)(Bt + k0);
            u32x4 g0;
            g0[0] = 1u;                                       // count=1, user descriptor
            g0[1] = lds_base;                                 // lds_addr
            g0[2] = (unsigned)ga;                             // global_addr[31:0]
            g0[3] = (unsigned)((ga >> 32) & 0x01FFFFFFu) | (2u << 30);  // addr[56:32] | type=2
            i32x8 g1;
            g1[0] = (int)((1u << 16)                          // data_size = 2 bytes
                        | (1u << 20)                          // pad_enable
                        | (3u << 22)                          // pad_interval: 16 DWORDs
                        | (3u << 25));                        // pad_amount: 4 DWORDs
            g1[1] = (int)((K & 0xFFFF) << 16);                // tensor_dim0[15:0] (bits 63:48)
            g1[2] = (int)(((K >> 16) & 0xFFFF) | ((N & 0xFFFF) << 16));  // dim0 hi | dim1 lo
            g1[3] = (int)(((N >> 16) & 0xFFFF) | (32u << 16));           // dim1 hi | tile_dim0=32
            g1[4] = (int)(N & 0xFFFF);                        // tile_dim1 = N (tile_dim2 = 0)
            g1[5] = K;                                        // tensor_dim0_stride lo32 (elements)
            g1[6] = 0;                                        // dim0_stride hi | dim1_stride lo
            g1[7] = 0;
            i32x4 g2 = {0, 0, 0, 0};
            i32x4 g3 = {0, 0, 0, 0};
#if defined(__clang_major__) && __clang_major__ >= 23
            i32x8 g4 = {};
            __builtin_amdgcn_tensor_load_to_lds(g0, g1, g2, g3, g4, 0);
#else
            __builtin_amdgcn_tensor_load_to_lds(g0, g1, g2, g3, 0);
#endif
        }
        // ---- stage A tile 64x32 (f32 -> f16), 8 halfs per thread, overlapped with the DMA
        {
            int row  = tid >> 2;
            int seg  = (tid & 3) * 8;
            int grow = m_block + row;
            _Float16* dst = &As[row * ASTR + seg];
            if (grow < M) {
                const float* src = A + (size_t)grow * K + k0 + seg;
#pragma unroll
                for (int j = 0; j < 8; ++j) dst[j] = (_Float16)src[j];
                if (k0 + BK < K) __builtin_prefetch(src + BK, 0, 0);   // global_prefetch_b8
            } else {
#pragma unroll
                for (int j = 0; j < 8; ++j) dst[j] = (_Float16)0.f;
            }
        }
        if (wave == 0) __builtin_amdgcn_s_wait_tensorcnt(0);
        __syncthreads();

        // ---- fragments per ISA 7.12.2 (wave32)
        // A 16x32: lanes 0-15 M=lane, K {0..7,16..23}; lanes 16-31 M=lane-16, K {8..15,24..31}
        // B 32x16: lanes 0-15 N=lane, K {0..7,16..23}; lanes 16-31 N=lane-16, K {8..15,24..31} (Bs is [n][k])
        v16h a0, a1, b0, b1, b2, b3;
#pragma unroll
        for (int e = 0; e < 16; ++e) {
            int ak = (e & 7) + ((e >> 3) * 16) + hi * 8;
            a0[e] = As[(wm +      hl) * ASTR + ak];
            a1[e] = As[(wm + 16 + hl) * ASTR + ak];
            int bk = (e & 7) + ((e >> 3) * 16) + hi * 8;
            b0[e] = Bs[(wn +      hl) * BSTR + bk];
            b1[e] = Bs[(wn + 16 + hl) * BSTR + bk];
            b2[e] = Bs[(wn + 32 + hl) * BSTR + bk];
            b3[e] = Bs[(wn + 48 + hl) * BSTR + bk];
        }
        c[0][0] = __builtin_amdgcn_wmma_f32_16x16x32_f16(false, a0, false, b0, (short)0, c[0][0], false, false);
        c[0][1] = __builtin_amdgcn_wmma_f32_16x16x32_f16(false, a0, false, b1, (short)0, c[0][1], false, false);
        c[0][2] = __builtin_amdgcn_wmma_f32_16x16x32_f16(false, a0, false, b2, (short)0, c[0][2], false, false);
        c[0][3] = __builtin_amdgcn_wmma_f32_16x16x32_f16(false, a0, false, b3, (short)0, c[0][3], false, false);
        c[1][0] = __builtin_amdgcn_wmma_f32_16x16x32_f16(false, a1, false, b0, (short)0, c[1][0], false, false);
        c[1][1] = __builtin_amdgcn_wmma_f32_16x16x32_f16(false, a1, false, b1, (short)0, c[1][1], false, false);
        c[1][2] = __builtin_amdgcn_wmma_f32_16x16x32_f16(false, a1, false, b2, (short)0, c[1][2], false, false);
        c[1][3] = __builtin_amdgcn_wmma_f32_16x16x32_f16(false, a1, false, b3, (short)0, c[1][3], false, false);
        __syncthreads();
    }

    // ---- store with bias. C/D: VGPR r -> M = r + 8*hi ; N = lane&15.
#pragma unroll
    for (int j = 0; j < 4; ++j) {
        int ncol = wn + j * 16 + hl;
        if (ncol < N) {
            float bb = bias[ncol];
#pragma unroll
            for (int r = 0; r < 8; ++r) {
                int m0 = m_block + wm + r + hi * 8;
                if (m0 < M)      C[(size_t)m0 * N + ncol]        = (OutT)(c[0][j][r] + bb);
                if (m0 + 16 < M) C[(size_t)(m0 + 16) * N + ncol] = (OutT)(c[1][j][r] + bb);
            }
        }
    }
}

// ---------------- weight prep: Wt[n][k] = f16(W[k][n]) ----------------
__global__ void transpose_to_f16(const float* __restrict__ W, _Float16* __restrict__ Wt, int K, int N)
{
    int t = blockIdx.x * blockDim.x + threadIdx.x;
    if (t >= K * N) return;
    int n = t / K, k = t - n * K;
    Wt[t] = (_Float16)W[(size_t)k * N + n];
}

// ---------------- softmax over 16 contiguous logits per (b,q,h) ----------------
__global__ void softmax16(float* __restrict__ attn, int total)
{
    int t = blockIdx.x * blockDim.x + threadIdx.x;
    if (t >= total) return;
    float* p = attn + (size_t)t * 16;
    float m = -1e30f;
#pragma unroll
    for (int i = 0; i < 16; ++i) m = fmaxf(m, p[i]);
    float s = 0.f;
#pragma unroll
    for (int i = 0; i < 16; ++i) { float e = __expf(p[i] - m); p[i] = e; s += e; }
    float inv = 1.f / s;
#pragma unroll
    for (int i = 0; i < 16; ++i) p[i] *= inv;
}

// ---------------- bilinear sampling: one wave32 per (b,q,h), lane = head channel ----------------
__global__ __launch_bounds__(256) void msda_sample(
    const _Float16* __restrict__ value_p,  // [BS, LV, 256] f16, channel = h*32+c
    const float* __restrict__ offs,        // [BS*LQ, 256]  = [.., h, l, p, 2]
    const float* __restrict__ attn,        // [BS*LQ, 128]  = [.., h, l, p]
    const float* __restrict__ refp,        // [BS*LQ, 4, 2]
    float* __restrict__ out_core)          // [BS*LQ, 256]
{
    const int lane = threadIdx.x & 31;
    const int wid  = blockIdx.x * 8 + (threadIdx.x >> 5);
    const int h    = wid & 7;
    const int bq   = wid >> 3;
    const int b    = bq / LQ_;

    const int HWs[4]    = {100, 50, 25, 13};
    const int starts[4] = {0, 10000, 12500, 13125};

    const float* offp = offs + (size_t)bq * 256 + h * 32;
    const float* awp  = attn + (size_t)bq * 128 + h * 16;
    const float* rp   = refp + (size_t)bq * 8;

    float acc = 0.f;
#pragma unroll
    for (int l = 0; l < 4; ++l) {
        const int Hl = HWs[l], Wl = HWs[l];
        const float rx = rp[l * 2 + 0];
        const float ry = rp[l * 2 + 1];
        const _Float16* vbase = value_p + ((size_t)b * LV_ + starts[l]) * 256 + h * 32 + lane;
#pragma unroll
        for (int p = 0; p < 4; ++p) {
            float ox = offp[l * 8 + p * 2 + 0];
            float oy = offp[l * 8 + p * 2 + 1];
            float x = (rx + ox / (float)Wl) * (float)Wl - 0.5f;   // align_corners=False
            float y = (ry + oy / (float)Hl) * (float)Hl - 0.5f;
            float x0f = floorf(x), y0f = floorf(y);
            int   x0 = (int)x0f,  y0 = (int)y0f;
            float lx = x - x0f,   ly = y - y0f;
            float aw = awp[l * 4 + p];
            float s = 0.f;
            if (y0 >= 0 && y0 < Hl) {
                if (x0 >= 0     && x0 < Wl)     s += (1.f - ly) * (1.f - lx) * (float)vbase[(size_t)(y0 * Wl + x0) * 256];
                if (x0 + 1 >= 0 && x0 + 1 < Wl) s += (1.f - ly) * lx         * (float)vbase[(size_t)(y0 * Wl + x0 + 1) * 256];
            }
            int y1 = y0 + 1;
            if (y1 >= 0 && y1 < Hl) {
                if (x0 >= 0     && x0 < Wl)     s += ly * (1.f - lx) * (float)vbase[(size_t)(y1 * Wl + x0) * 256];
                if (x0 + 1 >= 0 && x0 + 1 < Wl) s += ly * lx         * (float)vbase[(size_t)(y1 * Wl + x0 + 1) * 256];
            }
            acc += aw * s;
        }
    }
    out_core[(size_t)bq * 256 + h * 32 + lane] = acc;
}

// ---------------- launcher ----------------
extern "C" void kernel_launch(void* const* d_in, const int* in_sizes, int n_in,
                              void* d_out, int out_size, void* d_ws, size_t ws_size,
                              hipStream_t stream)
{
    const float* query  = (const float*)d_in[0];
    const float* refp   = (const float*)d_in[1];
    const float* value  = (const float*)d_in[2];
    // d_in[3] = value_spatial_shapes (int64, constant -> hardcoded)
    const float* W_off  = (const float*)d_in[4];
    const float* b_off  = (const float*)d_in[5];
    const float* W_attn = (const float*)d_in[6];
    const float* b_attn = (const float*)d_in[7];
    const float* W_val  = (const float*)d_in[8];
    const float* b_val  = (const float*)d_in[9];
    const float* W_out  = (const float*)d_in[10];
    const float* b_out  = (const float*)d_in[11];
    float* out = (float*)d_out;

    const int Mv = BS_ * LV_;   // 425408 value rows (divisible by 64)
    const int Mq = BS_ * LQ_;   // 9600 query rows

    // workspace carve-up (~232 MB)
    char* ws = (char*)d_ws;
    size_t off = 0;
    _Float16* value_p = (_Float16*)(ws + off); off += (size_t)Mv * 256 * 2;  // f16
    float*    offsbuf = (float*)(ws + off);    off += (size_t)Mq * 256 * 4;
    float*    attnbuf = (float*)(ws + off);    off += (size_t)Mq * 128 * 4;
    float*    core    = (float*)(ws + off);    off += (size_t)Mq * 256 * 4;
    _Float16* Wval_t  = (_Float16*)(ws + off); off += 256 * 256 * 2;
    _Float16* Woff_t  = (_Float16*)(ws + off); off += 256 * 256 * 2;
    _Float16* Wattn_t = (_Float16*)(ws + off); off += 256 * 128 * 2;
    _Float16* Wout_t  = (_Float16*)(ws + off); off += 256 * 256 * 2;

    dim3 blk(256);

    // 0) weight prep: transpose + f16 (tiny)
    transpose_to_f16<<<(256 * 256 + 255) / 256, blk, 0, stream>>>(W_val,  Wval_t,  256, 256);
    transpose_to_f16<<<(256 * 256 + 255) / 256, blk, 0, stream>>>(W_off,  Woff_t,  256, 256);
    transpose_to_f16<<<(256 * 128 + 255) / 256, blk, 0, stream>>>(W_attn, Wattn_t, 256, 128);
    transpose_to_f16<<<(256 * 256 + 255) / 256, blk, 0, stream>>>(W_out,  Wout_t,  256, 256);

    // 1) value projection (425408 x 256 x 256), output f16
    gemm_f16wmma<_Float16><<<dim3(Mv / BM), blk, 0, stream>>>(value, Wval_t, b_val, value_p, Mv, 256, 256);
    // 2) sampling offsets
    gemm_f16wmma<float><<<dim3(Mq / BM), blk, 0, stream>>>(query, Woff_t, b_off, offsbuf, Mq, 256, 256);
    // 3) attention logits (N=128; waves with wn>=128 are masked at store)
    gemm_f16wmma<float><<<dim3(Mq / BM), blk, 0, stream>>>(query, Wattn_t, b_attn, attnbuf, Mq, 128, 256);
    // 4) softmax over 16 per (b,q,h)
    softmax16<<<(Mq * 8 + 255) / 256, blk, 0, stream>>>(attnbuf, Mq * 8);
    // 5) deformable bilinear sampling -> core [Mq,256]
    msda_sample<<<Mq, blk, 0, stream>>>(value_p, offsbuf, attnbuf, refp, core);
    // 6) output projection -> d_out
    gemm_f16wmma<float><<<dim3(Mq / BM), blk, 0, stream>>>(core, Wout_t, b_out, out, Mq, 256, 256);
}
